// TemplatePairEmbedderMonomer_35476429865360
// MI455X (gfx1250) — matrix operand order repaired and checked by reading, not compiled, not run on hardware
//
#include <hip/hip_runtime.h>
#include <hip/hip_bf16.h>
#include <math.h>

// MI455X / gfx1250, wave32. Memory-bound kernel: 604MB of f32 output at
// 23.3 TB/s => ~26us floor. Feature tensor (88-wide) is synthesized in
// registers (one-hots + 5 scalars per pair) directly in WMMA A-operand
// layout; projection 88->64 done with v_wmma_f32_16x16x32_f16 (K padded to
// 96: 3 k-steps x 4 n-tiles = 12 WMMA per 16-pair tile).

typedef __attribute__((ext_vector_type(16))) _Float16 v16h;
typedef __attribute__((ext_vector_type(8)))  float    v8f;

#define NRES 768
#define NTPL 4
#define CIN  88
#define COUT 64

__global__ __launch_bounds__(256) void tpe_wmma_kernel(
    const float* __restrict__ pos,      // [4,768,37,3]
    const float* __restrict__ tpb,      // [4,768,3]
    const float* __restrict__ pbm,      // [4,768]
    const float* __restrict__ aam,      // [4,768,37]
    const int*   __restrict__ aatype,   // [4,768]
    const float* __restrict__ W,        // [64,88]
    const float* __restrict__ bias_in,  // [64]
    float* __restrict__ out)            // [4,768,768,64]
{
  const int lane = threadIdx.x & 31;
  const int wave = threadIdx.x >> 5;   // 0..7
  const int ln   = lane & 15;          // column / row-within-half
  const int g    = lane >> 4;          // half-wave group

  const int ti = blockIdx.x;           // 0..3071 = (t,i)
  const int t  = ti / NRES;
  const int i  = ti - t * NRES;
  const size_t ri = (size_t)t * NRES + i;

  // ---------------- preload B = W^T (f16) in WMMA B layout ----------------
  // B block (nt, s) is 32(K) x 16(N); lane holds column N = ln, K = s*32 + g*16 + e.
  v16h wb[4][3];
  float bias[4];
#pragma unroll
  for (int nt = 0; nt < 4; ++nt) {
    const int o = nt * 16 + ln;
    bias[nt] = bias_in[o];
#pragma unroll
    for (int s = 0; s < 3; ++s) {
#pragma unroll
      for (int e = 0; e < 16; ++e) {
        const int k = s * 32 + g * 16 + e;
        const float wv = (k < CIN) ? W[o * CIN + k] : 0.0f;
        wb[nt][s][e] = (_Float16)wv;
      }
    }
  }

  // ---------------- i-side (uniform across block) ----------------
  const float fm_i = aam[ri * 37 + 0] * aam[ri * 37 + 1] * aam[ri * 37 + 2];

  if (fm_i == 0.0f) {
    // act is identically zero for this row block -> output = bias.
    for (int itr = 0; itr < 6; ++itr) {
      const int jbase = (wave + 8 * itr) * 16;
      float* op = out + ((size_t)ti * NRES + jbase) * COUT;
#pragma unroll
      for (int nt = 0; nt < 4; ++nt)
#pragma unroll
        for (int r = 0; r < 8; ++r)
          __builtin_nontemporal_store(bias[nt],
              op + (size_t)(r + 8 * g) * COUT + nt * 16 + ln);
    }
    return;
  }

  const float pbm_i = pbm[ri];
  const int   kaai  = 62 + aatype[ri];
  const float pbix = tpb[ri * 3 + 0];
  const float pbiy = tpb[ri * 3 + 1];
  const float pbiz = tpb[ri * 3 + 2];

  // Backbone frame from N, CA, C (OpenFold make_transform_from_reference).
  const float* pb = pos + ri * 37 * 3;
  const float cax = pb[3], cay = pb[4], caz = pb[5];
  const float nx = pb[0] - cax, nyv = pb[1] - cay, nzv = pb[2] - caz;
  const float cx = pb[6] - cax, cyv = pb[7] - cay, czv = pb[8] - caz;
  const float EPS = 1e-6f;
  const float n1 = sqrtf(EPS + cx * cx + cyv * cyv);
  const float s1 = -cyv / n1, c1 = cx / n1;
  const float n2 = sqrtf(EPS + cx * cx + cyv * cyv + czv * czv);
  const float s2 = czv / n2, c2v = sqrtf(cx * cx + cyv * cyv) / n2;
  // c_rots = c2 @ c1
  const float r00 = c2v * c1, r01 = -c2v * s1, r02 = s2;
  const float r10 = s1,       r11 = c1,        r12 = 0.0f;
  const float r20 = -s2 * c1, r21 = s2 * s1,   r22 = c2v;
  // n_rot = c_rots @ n
  const float nry = r10 * nx + r11 * nyv + r12 * nzv;
  const float nrz = r20 * nx + r21 * nyv + r22 * nzv;
  const float n3 = sqrtf(EPS + nry * nry + nrz * nrz);
  const float sn = -nrz / n3, cn = nry / n3;
  // M = nr @ c_rots ; rigid_vec = M @ (ca_j - ca_i)  (== rots^T applied)
  const float m00 = r00, m01 = r01, m02 = r02;
  const float m10 = cn * r10 - sn * r20, m11 = cn * r11 - sn * r21, m12 = cn * r12 - sn * r22;
  const float m20 = sn * r10 + cn * r20, m21 = sn * r11 + cn * r21, m22 = sn * r12 + cn * r22;

  // ---------------- loop over 6 j-blocks of 16 ----------------
  for (int itr = 0; itr < 6; ++itr) {
    const int jbase = (wave + 8 * itr) * 16;
    const int j = jbase + ln;                 // this lane's pair row (2 lanes/row)
    const size_t rj = (size_t)t * NRES + j;

    const float fm_j = aam[rj * 37 + 0] * aam[rj * 37 + 1] * aam[rj * 37 + 2];
    const float fm2  = fm_i * fm_j;           // masks are 0/1
    const float pbm2f = pbm_i * pbm[rj] * fm2;
    const int   kaaj  = 40 + aatype[rj];

    // distogram bin (equivalent to d2>lower[b] && d2<upper[b] thresholds)
    const float dx = pbix - tpb[rj * 3 + 0];
    const float dy = pbiy - tpb[rj * 3 + 1];
    const float dz = pbiz - tpb[rj * 3 + 2];
    const float d2 = dx * dx + dy * dy + dz * dz;
    int bin = (int)floorf((sqrtf(d2) - 3.25f) * (1.0f / 1.25f));
    bin = bin > 38 ? 38 : bin;                // negative => no bin set

    // unit vector in frame i
    const float tx = pos[(rj * 37 + 1) * 3 + 0] - cax;
    const float ty = pos[(rj * 37 + 1) * 3 + 1] - cay;
    const float tz = pos[(rj * 37 + 1) * 3 + 2] - caz;
    const float rvx = m00 * tx + m01 * ty + m02 * tz;
    const float rvy = m10 * tx + m11 * ty + m12 * tz;
    const float rvz = m20 * tx + m21 * ty + m22 * tz;
    const float invd = fm2 * rsqrtf(EPS + rvx * rvx + rvy * rvy + rvz * rvz);
    const float uvx = rvx * invd, uvy = rvy * invd, uvz = rvz * invd;

    // Build A (16x96 f16, 3 k-steps) in WMMA A layout:
    // lane holds row M = ln; element e covers K = s*32 + 16g + e.
    v16h a0, a1, a2;
#pragma unroll
    for (int e = 0; e < 16; ++e) {
      const int kk = 16 * g + e;
      const int k0 = kk, k1 = 32 + kk, k2 = 64 + kk;
      float v0 = (k0 == bin) ? fm2 : 0.0f;                 // dgram bins 0..31
      float v1 = (k1 == bin) ? fm2 : 0.0f;                 // dgram bins 32..38
      v1 = (k1 == 39)   ? pbm2f : v1;                      // pb pair mask
      v1 = (k1 == kaaj) ? fm2   : v1;                      // oh_j start
      v1 = (k1 == kaai) ? fm2   : v1;                      // oh_i (62,63)
      float v2 = (k2 == kaai) ? fm2 : 0.0f;                // oh_i 64..83
      v2 = (k2 == 84) ? uvx : v2;
      v2 = (k2 == 85) ? uvy : v2;
      v2 = (k2 == 86) ? uvz : v2;
      v2 = (k2 == 87) ? fm2 : v2;                          // frame pair mask; k>=88 pad=0
      a0[e] = (_Float16)v0;
      a1[e] = (_Float16)v1;
      a2[e] = (_Float16)v2;
    }

    // 88->64 projection: 4 n-tiles x 3 k-steps of v_wmma_f32_16x16x32_f16
    v8f acc[4];
#pragma unroll
    for (int nt = 0; nt < 4; ++nt) {
      v8f c = {0.f, 0.f, 0.f, 0.f, 0.f, 0.f, 0.f, 0.f};
      c = __builtin_amdgcn_wmma_f32_16x16x32_f16(false, a0, false, wb[nt][0], (short)0, c, false, false);
      c = __builtin_amdgcn_wmma_f32_16x16x32_f16(false, a1, false, wb[nt][1], (short)0, c, false, false);
      c = __builtin_amdgcn_wmma_f32_16x16x32_f16(false, a2, false, wb[nt][2], (short)0, c, false, false);
      acc[nt] = c;
    }

    // Store C (16x16 f32 layout: VGPR r -> row r + 8g, lane -> column) + bias.
    float* op = out + ((size_t)ti * NRES + jbase) * COUT;
#pragma unroll
    for (int nt = 0; nt < 4; ++nt)
#pragma unroll
      for (int r = 0; r < 8; ++r)
        __builtin_nontemporal_store(acc[nt][r] + bias[nt],
            op + (size_t)(r + 8 * g) * COUT + nt * 16 + ln);
  }
}

extern "C" void kernel_launch(void* const* d_in, const int* in_sizes, int n_in,
                              void* d_out, int out_size, void* d_ws, size_t ws_size,
                              hipStream_t stream) {
  (void)in_sizes; (void)n_in; (void)out_size; (void)d_ws; (void)ws_size;
  const float* pos  = (const float*)d_in[0];
  const float* tpb  = (const float*)d_in[1];
  const float* pbmk = (const float*)d_in[2];
  const float* aam  = (const float*)d_in[3];
  const int*   aat  = (const int*)  d_in[4];
  const float* W    = (const float*)d_in[5];
  const float* bb   = (const float*)d_in[6];
  dim3 grid(NTPL * NRES);   // 3072 blocks, one per (t, i)
  dim3 block(256);          // 8 wave32 waves, each: 6 j-blocks of 16
  hipLaunchKernelGGL(tpe_wmma_kernel, grid, block, 0, stream,
                     pos, tpb, pbmk, aam, aat, W, bb, (float*)d_out);
}
